// LlamaAttention_63342177681912
// MI455X (gfx1250) — matrix operand order repaired and checked by reading, not compiled, not run on hardware
//
#include <hip/hip_runtime.h>

// ---------------------------------------------------------------------------
// LlamaAttention for MI455X (gfx1250).
// bf16 WMMA everywhere (fp32 accum), TDM (tensor_load_to_lds) GEMM staging,
// 8-wave shared-KV flash attention.
// HIDDEN=2048, H=32, HKV=8, HD=64, B=2, S=2048.
// Workspace requirement: ~132 MB.
// ---------------------------------------------------------------------------

#define HIDDEN   2048
#define NHEADS   32
#define NKVHEADS 8
#define HEADDIM  64
#define BATCH    2
#define SEQ      2048

typedef __attribute__((ext_vector_type(16))) __bf16     v16bf;
typedef __attribute__((ext_vector_type(8)))  float      v8f;
typedef __attribute__((ext_vector_type(8)))  unsigned   v8u;
typedef __attribute__((ext_vector_type(4)))  unsigned   v4u;

// fp32 -> bf16, round-to-nearest-even
__device__ __forceinline__ unsigned short f2bf(float x) {
  unsigned u = __float_as_uint(x);
  u += 0x7FFFu + ((u >> 16) & 1u);
  return (unsigned short)(u >> 16);
}

// A-matrix fragment (16x32 bf16, MxK), rowbase = start of this lane's row.
// ISA layout: lanes 0-15: K=0..7,16..23 ; lanes 16-31: K=8..15,24..31.
__device__ __forceinline__ v16bf load_frag_a(const unsigned short* rowbase, int lane) {
  const int koff = (lane >= 16) ? 8 : 0;
  v8u u;
#pragma unroll
  for (int i = 0; i < 4; ++i) {
    u[i]     = *(const unsigned*)(rowbase + koff + 2 * i);
    u[i + 4] = *(const unsigned*)(rowbase + koff + 16 + 2 * i);
  }
  return __builtin_bit_cast(v16bf, u);
}

// B-matrix fragment (32x16 bf16, KxN), data stored K-contiguous per N row.
// Lane holds N = lane&15; lanes 0-15: K=0..7,16..23 ; lanes 16-31: K=8..15,24..31.
__device__ __forceinline__ v16bf load_frag_b(const unsigned short* nrowbase, int lane) {
  const int koff = (lane >= 16) ? 8 : 0;
  v8u u;
#pragma unroll
  for (int i = 0; i < 8; ++i)
    u[i] = *(const unsigned*)(nrowbase + koff + 2 * (i & 3) + ((i >> 2) << 4));
  return __builtin_bit_cast(v16bf, u);
}

__device__ __forceinline__ v8f wmma_bf16(v16bf a, v16bf b, v8f c) {
  return __builtin_amdgcn_wmma_f32_16x16x32_bf16(false, a, false, b, (short)0, c, false, false);
}

// ---------------------------------------------------------------------------
// Tensor Data Mover: 2D bf16 tile (tileD0 contiguous elems x tileD1 rows)
// global -> LDS. D# built per CDNA5 ISA sec. 8.3/8.4. Issue once per wave.
// ---------------------------------------------------------------------------
__device__ __forceinline__ void tdm_load_2d_bf16(unsigned ldsOff, const void* gptr,
                                                 unsigned tileD0, unsigned tileD1,
                                                 unsigned long long strideElems) {
  unsigned long long ga = (unsigned long long)(uintptr_t)gptr;
  v4u g0;
  g0[0] = 1u;                                    // count=1, user descriptor
  g0[1] = ldsOff;                                // lds_addr (bytes)
  g0[2] = (unsigned)(ga & 0xFFFFFFFFu);          // global_addr[31:0]
  g0[3] = (unsigned)((ga >> 32) & 0x01FFFFFFu)   // global_addr[56:32]
          | (2u << 30);                          // type = 2 (image)
  const unsigned td0 = 0x40000000u;              // huge tensor dims: tiles never OOB
  const unsigned td1 = 0x40000000u;
  v8u g1;
  g1[0] = 0x00010000u;                           // data_size=1 -> 2 bytes/elem
  g1[1] = (td0 & 0xFFFFu) << 16;                 // tensor_dim0[15:0]
  g1[2] = (td0 >> 16) | ((td1 & 0xFFFFu) << 16); // tensor_dim0[31:16] | tensor_dim1[15:0]
  g1[3] = (td1 >> 16) | (tileD0 << 16);          // tensor_dim1[31:16] | tile_dim0
  g1[4] = tileD1;                                // tile_dim1 (tile_dim2 = 0)
  g1[5] = (unsigned)(strideElems & 0xFFFFFFFFu); // tensor_dim0_stride[31:0]
  g1[6] = (unsigned)((strideElems >> 32) & 0xFFFFu);
  g1[7] = 0;
  v4u gz = {0u, 0u, 0u, 0u};                     // groups 2/3 unused (2D)
  asm volatile("tensor_load_to_lds %0, %1, %2, %3"
               :: "s"(g0), "s"(g1), "s"(gz), "s"(gz)
               : "memory");
}

// ---------------------------------------------------------------------------
// GEMM  Y[M,N] = A[M,K] @ Bt[N,K]^T, bf16 in, fp32 out.
// 256 threads / 8 waves; block tile 64x64, k-step 64; wave owns 16x32 C.
// TDM stages both tiles; 4 WMMAs per wave per k-step.
// ---------------------------------------------------------------------------
__global__ __launch_bounds__(256) void gemm_bf16_tdm(const unsigned short* __restrict__ A,
                                                     const unsigned short* __restrict__ Bt,
                                                     float* __restrict__ Y,
                                                     int M, int K, int N,
                                                     int nHeads, int headMajor) {
  __shared__ unsigned short As[64][64];  // [m][k]
  __shared__ unsigned short Bs[64][64];  // [n][k]

  const int tid  = threadIdx.x;
  const int lane = tid & 31;
  const int wave = tid >> 5;
  const int wm   = wave >> 1;                 // 0..3
  const int wn   = wave & 1;                  // 0..1
  const int m0   = (int)blockIdx.x * 64;
  const int n0   = (int)blockIdx.y * 64;
  const int row  = lane & 15;

  const unsigned ldsA = (unsigned)(uintptr_t)&As[0][0];
  const unsigned ldsB = (unsigned)(uintptr_t)&Bs[0][0];

  v8f c0 = {}, c1 = {};

  for (int kb = 0; kb < K; kb += 64) {
    __syncthreads();                           // prior frag reads done
    if (tid < 32) {                            // wave 0 drives the TDM
      tdm_load_2d_bf16(ldsA, A  + ((size_t)m0 * K + kb), 64, 64, (unsigned long long)K);
      tdm_load_2d_bf16(ldsB, Bt + ((size_t)n0 * K + kb), 64, 64, (unsigned long long)K);
      __builtin_amdgcn_s_wait_tensorcnt(0);
    }
    __syncthreads();                           // tiles visible to all waves

#pragma unroll
    for (int kc = 0; kc < 64; kc += 32) {
      v16bf a  = load_frag_a(&As[wm * 16 + row][kc], lane);
      v16bf b0 = load_frag_b(&Bs[wn * 32 + row][kc], lane);
      v16bf b1 = load_frag_b(&Bs[wn * 32 + 16 + row][kc], lane);
      c0 = wmma_bf16(a, b0, c0);
      c1 = wmma_bf16(a, b1, c1);
    }
  }

  // C layout: lane l, reg r -> m = r + 8*(l>=16), n = l&15
  if (headMajor) {
#pragma unroll
    for (int r = 0; r < 8; ++r) {
      int m  = m0 + wm * 16 + r + 8 * (lane >> 4);
      int b_ = m / SEQ, s = m % SEQ;
      int n  = n0 + wn * 32 + row;
      int h0 = n >> 6, d0 = n & 63;
      int h1 = (n + 16) >> 6, d1 = (n + 16) & 63;
      Y[(((size_t)b_ * nHeads + h0) * SEQ + s) * HEADDIM + d0] = c0[r];
      Y[(((size_t)b_ * nHeads + h1) * SEQ + s) * HEADDIM + d1] = c1[r];
    }
  } else {
#pragma unroll
    for (int r = 0; r < 8; ++r) {
      int m = m0 + wm * 16 + r + 8 * (lane >> 4);
      int n = n0 + wn * 32 + row;
      Y[(size_t)m * N + n]      = c0[r];
      Y[(size_t)m * N + n + 16] = c1[r];
    }
  }
}

// ---------------------------------------------------------------------------
// Elementwise converters
// ---------------------------------------------------------------------------
__global__ void cvt_f32_to_bf16(const float* __restrict__ src,
                                unsigned short* __restrict__ dst, long long n) {
  long long i = ((long long)blockIdx.x * blockDim.x + threadIdx.x) * 4;
  if (i >= n) return;
  unsigned* d = (unsigned*)(dst + i);
  d[0] = (unsigned)f2bf(src[i])     | ((unsigned)f2bf(src[i + 1]) << 16);
  d[1] = (unsigned)f2bf(src[i + 2]) | ((unsigned)f2bf(src[i + 3]) << 16);
}

// Wt[n*K + k] = bf16(W[k*N + n]); grid (K/256, N)
__global__ void transpose_cvt_bf16(const float* __restrict__ W,
                                   unsigned short* __restrict__ Wt, int K, int N) {
  int k = (int)blockIdx.x * 256 + threadIdx.x;
  int n = (int)blockIdx.y;
  Wt[(size_t)n * K + k] = f2bf(W[(size_t)k * N + n]);
}

// RoPE fp32 [B,nH,S,64] -> bf16, positions = arange(S)
__global__ void rope_pack_bf16(const float* __restrict__ src,
                               unsigned short* __restrict__ dst, long long totalPairs) {
  long long idx = (long long)blockIdx.x * blockDim.x + threadIdx.x;
  if (idx >= totalPairs) return;
  int d = (int)(idx & 31);
  long long rowi = idx >> 5;
  int pos = (int)(rowi % SEQ);
  float freq = __powf(10000.0f, -(float)(2 * d) * (1.0f / 64.0f));
  float ang  = (float)pos * freq;
  float cs = cosf(ang), sn = sinf(ang);
  const float* p = src + rowi * HEADDIM;
  float x1 = p[d], x2 = p[d + 32];
  dst[rowi * HEADDIM + d]      = f2bf(x1 * cs - x2 * sn);
  dst[rowi * HEADDIM + d + 32] = f2bf(x2 * cs + x1 * sn);
}

// ---------------------------------------------------------------------------
// Flash attention: 256-thread block = 8 waves = 128 queries of one (b,h).
// All waves share one 32-key K/V LDS tile (8x less K/V HBM traffic).
// bf16 in, bf16 out ([B,S,2048] head-concat, feeds final GEMM directly).
// ---------------------------------------------------------------------------
__global__ __launch_bounds__(256) void flash_attn(const unsigned short* __restrict__ Q,
                                                  const unsigned short* __restrict__ Kg,
                                                  const unsigned short* __restrict__ Vg,
                                                  unsigned short* __restrict__ O) {
  __shared__ unsigned short Ks[32][64];       // [key][d]
  __shared__ unsigned short Vt[64][32];       // [d][key]
  __shared__ unsigned short Ps[8][16][32];    // per-wave P tile

  const int tid  = threadIdx.x;
  const int lane = tid & 31;
  const int wave = tid >> 5;
  const int q0b  = (int)blockIdx.x * 128;
  const int h    = (int)blockIdx.y;
  const int b    = (int)blockIdx.z;
  const int q0   = q0b + wave * 16;
  const int hk   = h >> 2;
  const int row  = lane & 15;
  const int half = lane >> 4;

  const unsigned short* Qb = Q  + (((size_t)b * NHEADS   + h ) * SEQ + q0) * HEADDIM;
  const unsigned short* Kb = Kg + (((size_t)b * NKVHEADS + hk) * SEQ) * HEADDIM;
  const unsigned short* Vb = Vg + (((size_t)b * NKVHEADS + hk) * SEQ) * HEADDIM;

  // Q fragments loaded straight from global bf16 (once)
  const v16bf qa0 = load_frag_a(Qb + (size_t)row * HEADDIM, lane);
  const v16bf qa1 = load_frag_a(Qb + (size_t)row * HEADDIM + 32, lane);

  float mi[8], li[8];
  v8f o0 = {}, o1 = {}, o2 = {}, o3 = {};
#pragma unroll
  for (int r = 0; r < 8; ++r) { mi[r] = -3.0e38f; li[r] = 0.0f; }

  const float scl = 0.125f;      // 1/sqrt(64)
  const int nkeys = q0b + 128;   // causal span of the whole block

  for (int kb = 0; kb < nkeys; kb += 32) {
    __syncthreads();
    for (int i = tid; i < 32 * 64; i += 256) {     // stage K and V^T
      int kr = i >> 6, d = i & 63;
      Ks[kr][d] = Kb[(size_t)(kb + kr) * HEADDIM + d];
      Vt[d][kr] = Vb[(size_t)(kb + kr) * HEADDIM + d];
    }
    __syncthreads();

    // scores S[16x32]: two 16x16 C tiles, head-dim 64 chained
    v8f s0 = {}, s1 = {};
    s0 = wmma_bf16(qa0, load_frag_b(&Ks[row][0],       lane), s0);
    s0 = wmma_bf16(qa1, load_frag_b(&Ks[row][32],      lane), s0);
    s1 = wmma_bf16(qa0, load_frag_b(&Ks[16 + row][0],  lane), s1);
    s1 = wmma_bf16(qa1, load_frag_b(&Ks[16 + row][32], lane), s1);

    float al[8];
#pragma unroll
    for (int r = 0; r < 8; ++r) {
      int m    = r + 8 * half;
      int qidx = q0 + m;
      float v0 = s0[r] * scl;
      float v1 = s1[r] * scl;
      if (kb + row      > qidx) v0 = -3.0e38f;
      if (kb + 16 + row > qidx) v1 = -3.0e38f;

      float rv = fmaxf(v0, v1);                 // row max over 16 lanes (same m)
      rv = fmaxf(rv, __shfl_xor(rv, 1, 32));
      rv = fmaxf(rv, __shfl_xor(rv, 2, 32));
      rv = fmaxf(rv, __shfl_xor(rv, 4, 32));
      rv = fmaxf(rv, __shfl_xor(rv, 8, 32));
      float mnew = fmaxf(mi[r], rv);
      al[r] = __expf(mi[r] - mnew);

      float p0 = __expf(v0 - mnew);
      float p1 = __expf(v1 - mnew);
      float rs = p0 + p1;
      rs += __shfl_xor(rs, 1, 32);
      rs += __shfl_xor(rs, 2, 32);
      rs += __shfl_xor(rs, 4, 32);
      rs += __shfl_xor(rs, 8, 32);
      li[r] = li[r] * al[r] + rs;
      mi[r] = mnew;

      Ps[wave][m][row]      = f2bf(p0);
      Ps[wave][m][16 + row] = f2bf(p1);
    }
#pragma unroll
    for (int r = 0; r < 8; ++r) {
      o0[r] *= al[r]; o1[r] *= al[r]; o2[r] *= al[r]; o3[r] *= al[r];
    }
    __syncthreads();

    v16bf pf = load_frag_a(&Ps[wave][row][0], lane);              // 16x32 over keys
    o0 = wmma_bf16(pf, load_frag_b(&Vt[row][0],      lane), o0);  // d  0..15
    o1 = wmma_bf16(pf, load_frag_b(&Vt[16 + row][0], lane), o1);  // d 16..31
    o2 = wmma_bf16(pf, load_frag_b(&Vt[32 + row][0], lane), o2);  // d 32..47
    o3 = wmma_bf16(pf, load_frag_b(&Vt[48 + row][0], lane), o3);  // d 48..63
  }

  unsigned short* Ob = O + ((size_t)b * SEQ + q0) * HIDDEN + h * HEADDIM;
#pragma unroll
  for (int r = 0; r < 8; ++r) {
    int m = r + 8 * half;
    float inv = 1.0f / li[r];
    Ob[(size_t)m * HIDDEN + row]      = f2bf(o0[r] * inv);
    Ob[(size_t)m * HIDDEN + 16 + row] = f2bf(o1[r] * inv);
    Ob[(size_t)m * HIDDEN + 32 + row] = f2bf(o2[r] * inv);
    Ob[(size_t)m * HIDDEN + 48 + row] = f2bf(o3[r] * inv);
  }
}

// ---------------------------------------------------------------------------
// Launch sequence
// ---------------------------------------------------------------------------
extern "C" void kernel_launch(void* const* d_in, const int* in_sizes, int n_in,
                              void* d_out, int out_size, void* d_ws, size_t ws_size,
                              hipStream_t stream) {
  const float* X  = (const float*)d_in[0];
  const float* Wq = (const float*)d_in[1];
  const float* Wk = (const float*)d_in[2];
  const float* Wv = (const float*)d_in[3];
  const float* Wo = (const float*)d_in[4];
  // d_in[5] = position_ids == arange(S): handled analytically.

  const int M = BATCH * SEQ;                 // 4096
  const size_t nX  = (size_t)M * HIDDEN;     // 8.4M
  const size_t nQ  = (size_t)BATCH * NHEADS   * SEQ * HEADDIM;   // 8.4M
  const size_t nKV = (size_t)BATCH * NKVHEADS * SEQ * HEADDIM;   // 2.1M
  const size_t nWq = (size_t)HIDDEN * HIDDEN;                    // 4.2M
  const size_t nWk = (size_t)HIDDEN * (NKVHEADS * HEADDIM);      // 1.05M

  // fp32 scratch
  float* Qf = (float*)d_ws;
  float* Kf = Qf + nQ;
  float* Vf = Kf + nKV;
  // bf16 scratch
  unsigned short* Xb  = (unsigned short*)(Vf + nKV);
  unsigned short* Wqt = Xb  + nX;
  unsigned short* Wkt = Wqt + nWq;
  unsigned short* Wvt = Wkt + nWk;
  unsigned short* Wot = Wvt + nWk;
  unsigned short* Qbf = Wot + nWq;
  unsigned short* Kbf = Qbf + nQ;
  unsigned short* Vbf = Kbf + nKV;
  unsigned short* Abf = Vbf + nKV;

  // 1) convert inputs to bf16 (weights transposed to [N][K])
  cvt_f32_to_bf16<<<(unsigned)((nX / 4 + 255) / 256), 256, 0, stream>>>(X, Xb, (long long)nX);
  transpose_cvt_bf16<<<dim3(HIDDEN / 256, HIDDEN), 256, 0, stream>>>(Wq, Wqt, HIDDEN, HIDDEN);
  transpose_cvt_bf16<<<dim3(HIDDEN / 256, NKVHEADS * HEADDIM), 256, 0, stream>>>(Wk, Wkt, HIDDEN, NKVHEADS * HEADDIM);
  transpose_cvt_bf16<<<dim3(HIDDEN / 256, NKVHEADS * HEADDIM), 256, 0, stream>>>(Wv, Wvt, HIDDEN, NKVHEADS * HEADDIM);
  transpose_cvt_bf16<<<dim3(HIDDEN / 256, HIDDEN), 256, 0, stream>>>(Wo, Wot, HIDDEN, HIDDEN);

  // 2) QKV projections (fp32 out, head-major [B,H,S,64])
  gemm_bf16_tdm<<<dim3(M / 64, HIDDEN / 64), 256, 0, stream>>>(
      Xb, Wqt, Qf, M, HIDDEN, HIDDEN, NHEADS, 1);
  gemm_bf16_tdm<<<dim3(M / 64, (NKVHEADS * HEADDIM) / 64), 256, 0, stream>>>(
      Xb, Wkt, Kf, M, HIDDEN, NKVHEADS * HEADDIM, NKVHEADS, 1);
  gemm_bf16_tdm<<<dim3(M / 64, (NKVHEADS * HEADDIM) / 64), 256, 0, stream>>>(
      Xb, Wvt, Vf, M, HIDDEN, NKVHEADS * HEADDIM, NKVHEADS, 1);

  // 3) RoPE (fp32 -> bf16) on Q,K ; plain convert V
  {
    long long pq = (long long)nQ / 2;
    rope_pack_bf16<<<(unsigned)((pq + 255) / 256), 256, 0, stream>>>(Qf, Qbf, pq);
    long long pk = (long long)nKV / 2;
    rope_pack_bf16<<<(unsigned)((pk + 255) / 256), 256, 0, stream>>>(Kf, Kbf, pk);
    cvt_f32_to_bf16<<<(unsigned)((nKV / 4 + 255) / 256), 256, 0, stream>>>(Vf, Vbf, (long long)nKV);
  }

  // 4) causal GQA flash attention -> bf16 [B,S,2048]
  flash_attn<<<dim3(SEQ / 128, NHEADS, BATCH), 256, 0, stream>>>(Qbf, Kbf, Vbf, Abf);

  // 5) output projection -> d_out fp32 [B,S,2048]
  gemm_bf16_tdm<<<dim3(M / 64, HIDDEN / 64), 256, 0, stream>>>(
      Abf, Wot, (float*)d_out, M, HIDDEN, HIDDEN, 1, 0);
}
